// LSTM_8710193677080
// MI455X (gfx1250) — compile-verified, matrix operand order checked by reading, not run on hardware
//
#include <hip/hip_runtime.h>
#include <stdint.h>
#include <stddef.h>

#define TT 8192
#define HH 1024
#define GG 4096   /* 4*H */
#define KK 1024   /* reduction dim (I == H == 1024) */
#define NWG 64
#define WROW 1028 /* padded LDS row stride (floats) to rotate banks */

typedef __bf16 v16bf __attribute__((ext_vector_type(16)));
typedef float  v8f   __attribute__((ext_vector_type(8)));

union FragU { v16bf v; uint4 q[2]; };

__device__ __forceinline__ v16bf load_frag(const unsigned short* p) {
  FragU u;
  u.q[0] = *(const uint4*)(p);        // K elems e0..e7  (contiguous 16B)
  u.q[1] = *(const uint4*)(p + 16);   // K elems e8..e15 (contiguous 16B)
  return u.v;
}

__device__ __forceinline__ unsigned short f32_to_bf16_rn(float f) {
  unsigned u = __float_as_uint(f);
  return (unsigned short)((u + 0x7FFFu + ((u >> 16) & 1u)) >> 16);
}

#define WMMA_BF16(A, B, C) \
  __builtin_amdgcn_wmma_f32_16x16x32_bf16(false, (A), false, (B), (short)0, (C), false, false)

// ---------------------------------------------------------------- split f32 -> bf16 hi/lo
__global__ void split_bf16_kernel(const float* __restrict__ in,
                                  unsigned short* __restrict__ hi,
                                  unsigned short* __restrict__ lo, int n) {
  int idx = blockIdx.x * blockDim.x + threadIdx.x;
  int stride = gridDim.x * blockDim.x;
  for (; idx < n; idx += stride) {
    float f = in[idx];
    unsigned short h = f32_to_bf16_rn(f);
    float hf = __uint_as_float(((unsigned)h) << 16);
    hi[idx] = h;
    lo[idx] = f32_to_bf16_rn(f - hf);
  }
}

// ---------------------------------------------------------------- WMMA GEMM:
// C[M,4096] = A[M,1024] @ W[4096,1024]^T + b0 + b1, bf16 hi/lo 3-product split.
// Each wave owns a 32x32 macro-tile (2x2 WMMA tiles): 8 fragment loads feed 12 WMMAs/k-step.
__global__ __launch_bounds__(256) void wmma_gemm_bias_kernel(
    const unsigned short* __restrict__ Ahi, const unsigned short* __restrict__ Alo,
    const unsigned short* __restrict__ Bhi, const unsigned short* __restrict__ Blo,
    const float* __restrict__ bias0, const float* __restrict__ bias1,
    float* __restrict__ C) {
  const int N2 = GG / 32;  // macro-tiles along N
  int wave = blockIdx.x * (blockDim.x >> 5) + (threadIdx.x >> 5);
  int lane = threadIdx.x & 31;
  int nt = wave % N2, mt = wave / N2;
  int row0 = mt << 5, col0 = nt << 5;
  int r = lane & 15, half = lane >> 4;
  int c0 = col0 + r, c1 = col0 + 16 + r;

  float bv0 = bias0[c0] + bias1[c0];
  float bv1 = bias0[c1] + bias1[c1];
  v8f a00, a01, a10, a11;
#pragma unroll
  for (int j = 0; j < 8; ++j) { a00[j] = bv0; a10[j] = bv0; a01[j] = bv1; a11[j] = bv1; }

  const unsigned short* pah0 = Ahi + (size_t)(row0 + r) * KK + 8 * half;
  const unsigned short* pal0 = Alo + (size_t)(row0 + r) * KK + 8 * half;
  const unsigned short* pah1 = Ahi + (size_t)(row0 + 16 + r) * KK + 8 * half;
  const unsigned short* pal1 = Alo + (size_t)(row0 + 16 + r) * KK + 8 * half;
  const unsigned short* pbh0 = Bhi + (size_t)c0 * KK + 8 * half;
  const unsigned short* pbl0 = Blo + (size_t)c0 * KK + 8 * half;
  const unsigned short* pbh1 = Bhi + (size_t)c1 * KK + 8 * half;
  const unsigned short* pbl1 = Blo + (size_t)c1 * KK + 8 * half;

  for (int k0 = 0; k0 < KK; k0 += 32) {
    v16bf ah0 = load_frag(pah0 + k0);
    v16bf al0 = load_frag(pal0 + k0);
    v16bf ah1 = load_frag(pah1 + k0);
    v16bf al1 = load_frag(pal1 + k0);
    v16bf bh0 = load_frag(pbh0 + k0);
    v16bf bl0 = load_frag(pbl0 + k0);
    v16bf bh1 = load_frag(pbh1 + k0);
    v16bf bl1 = load_frag(pbl1 + k0);

    a00 = WMMA_BF16(ah0, bh0, a00);
    a00 = WMMA_BF16(ah0, bl0, a00);
    a00 = WMMA_BF16(al0, bh0, a00);

    a01 = WMMA_BF16(ah0, bh1, a01);
    a01 = WMMA_BF16(ah0, bl1, a01);
    a01 = WMMA_BF16(al0, bh1, a01);

    a10 = WMMA_BF16(ah1, bh0, a10);
    a10 = WMMA_BF16(ah1, bl0, a10);
    a10 = WMMA_BF16(al1, bh0, a10);

    a11 = WMMA_BF16(ah1, bh1, a11);
    a11 = WMMA_BF16(ah1, bl1, a11);
    a11 = WMMA_BF16(al1, bh1, a11);
  }

#pragma unroll
  for (int j = 0; j < 8; ++j) {
    size_t rA = (size_t)(row0 + j + 8 * half);
    size_t rB = rA + 16;
    C[rA * GG + c0] = a00[j];
    C[rA * GG + c1] = a01[j];
    C[rB * GG + c0] = a10[j];
    C[rB * GG + c1] = a11[j];
  }
}

// ---------------------------------------------------------------- grid barrier (generation based)
__device__ __forceinline__ void grid_barrier(unsigned* cnt, unsigned* gen) {
  __syncthreads();
  if (threadIdx.x == 0) {
    __threadfence();
    unsigned g = __hip_atomic_load(gen, __ATOMIC_RELAXED, __HIP_MEMORY_SCOPE_AGENT);
    unsigned arrived = atomicAdd(cnt, 1u);
    if (arrived == NWG - 1u) {
      *cnt = 0u;
      __threadfence();
      atomicAdd(gen, 1u);
    } else {
      while (__hip_atomic_load(gen, __ATOMIC_RELAXED, __HIP_MEMORY_SCOPE_AGENT) == g)
        __builtin_amdgcn_s_sleep(2);
    }
    __threadfence();
  }
  __syncthreads();
}

__device__ __forceinline__ float sigf(float x) { return 1.0f / (1.0f + expf(-x)); }

// ---------------------------------------------------------------- persistent LSTM recurrence
// 64 WGs x 256 threads. WG w owns h[w*16 .. w*16+16) and the 64 W_hh rows
// {g*1024 + w*16 + j} (g=0..3 gates, j=0..15), preloaded into LDS (257 KB < 320 KB WGP LDS).
__global__ __launch_bounds__(256) void lstm_recur_kernel(
    const float* __restrict__ xg,   // [T, 4096] precomputed x@W_ih^T + biases
    const float* __restrict__ Whh,  // [4096, 1024]
    float* __restrict__ y,          // [T, 1024] layer output (nullptr to skip)
    float* __restrict__ hbuf,       // [1024] global h exchange
    float* __restrict__ dst,        // final-h destination (nullptr to skip)
    unsigned* __restrict__ cnt, unsigned* __restrict__ gen, int T) {
  extern __shared__ float lds[];
  float* wlds = lds;                 // 64 * WROW
  float* hlds = lds + 64 * WROW;     // 1024
  float* dots = hlds + HH;           // 64
  float* cst  = dots + 64;           // 16 (cell state slice)

  const int tid = threadIdx.x;
  const int wg  = blockIdx.x;

  // preload W_hh slice (strided gate rows) into LDS, float4-wide
  for (int idx = tid; idx < 64 * (HH / 4); idx += 256) {
    int rr = idx >> 8;
    int k4 = idx & 255;
    int grow = ((rr >> 4) * HH) + (wg << 4) + (rr & 15);
    *(float4*)&wlds[rr * WROW + (k4 << 2)] =
        *(const float4*)&Whh[(size_t)grow * HH + (k4 << 2)];
  }
  if (tid < 16) {
    cst[tid] = 0.0f;
    hbuf[(wg << 4) + tid] = 0.0f;   // h0 = 0
  }
  grid_barrier(cnt, gen);

  const int r = tid >> 2, q = tid & 3;          // 4 threads per gate row
  const float4* w4 = (const float4*)(wlds + r * WROW + q * 256);
  const float4* h4 = (const float4*)(hlds + q * 256);

  for (int t = 0; t < T; ++t) {
    ((float4*)hlds)[tid] = ((const float4*)hbuf)[tid];   // 256 threads x 16B = 1024 floats
    if (tid < 16 && t + 1 < T) {
      // warm L0/L2 for next step's gate pre-activations (global_prefetch_b8)
      const float* nx = xg + (size_t)(t + 1) * GG + (wg << 4) + tid;
      __builtin_prefetch(nx, 0, 1);
      __builtin_prefetch(nx + HH, 0, 1);
      __builtin_prefetch(nx + 2 * HH, 0, 1);
      __builtin_prefetch(nx + 3 * HH, 0, 1);
    }
    __syncthreads();

    float s = 0.0f;
#pragma unroll 8
    for (int i = 0; i < 64; ++i) {
      float4 a = w4[i], b = h4[i];
      s = fmaf(a.x, b.x, s); s = fmaf(a.y, b.y, s);
      s = fmaf(a.z, b.z, s); s = fmaf(a.w, b.w, s);
    }
    s += __shfl_xor(s, 1, 32);
    s += __shfl_xor(s, 2, 32);
    if (q == 0) dots[r] = s;
    __syncthreads();

    if (tid < 16) {
      const float* xgt = xg + (size_t)t * GG + (wg << 4) + tid;
      float gi = xgt[0]      + dots[tid];
      float gf = xgt[HH]     + dots[16 + tid];
      float gc = xgt[2 * HH] + dots[32 + tid];
      float go = xgt[3 * HH] + dots[48 + tid];
      float c = sigf(gf) * cst[tid] + sigf(gi) * tanhf(gc);
      float h = sigf(go) * tanhf(c);
      cst[tid] = c;
      hbuf[(wg << 4) + tid] = h;
      if (y) y[(size_t)t * HH + (wg << 4) + tid] = h;
      if (dst && t == T - 1) dst[(wg << 4) + tid] = h;
    }
    __threadfence();
    grid_barrier(cnt, gen);
  }
}

// ---------------------------------------------------------------- host
extern "C" void kernel_launch(void* const* d_in, const int* in_sizes, int n_in,
                              void* d_out, int out_size, void* d_ws, size_t ws_size,
                              hipStream_t stream) {
  const float* x     = (const float*)d_in[0];
  const float* W_ih0 = (const float*)d_in[1];
  const float* W_hh0 = (const float*)d_in[2];
  const float* b_ih0 = (const float*)d_in[3];
  const float* b_hh0 = (const float*)d_in[4];
  const float* W_ih1 = (const float*)d_in[5];
  const float* W_hh1 = (const float*)d_in[6];
  const float* b_ih1 = (const float*)d_in[7];
  const float* b_hh1 = (const float*)d_in[8];
  float* out = (float*)d_out;

  char* ws = (char*)d_ws;
  size_t off = 0;
  auto bump = [&](size_t b) { size_t o = off; off += (b + 255) & ~(size_t)255; return o; };
  size_t xg_off  = bump((size_t)TT * GG * 4);   // 128 MB, reused per layer
  size_t xhi_off = bump((size_t)TT * KK * 2);
  size_t xlo_off = bump((size_t)TT * KK * 2);
  size_t whi_off = bump((size_t)GG * KK * 2);
  size_t wlo_off = bump((size_t)GG * KK * 2);
  size_t y0_off  = bump((size_t)TT * HH * 4);
  size_t h_off   = bump((size_t)HH * 4);
  size_t bar_off = bump(256);
  (void)ws_size; (void)in_sizes; (void)n_in; (void)out_size;

  float* xg = (float*)(ws + xg_off);
  unsigned short* xhi = (unsigned short*)(ws + xhi_off);
  unsigned short* xlo = (unsigned short*)(ws + xlo_off);
  unsigned short* whi = (unsigned short*)(ws + whi_off);
  unsigned short* wlo = (unsigned short*)(ws + wlo_off);
  float* y0   = (float*)(ws + y0_off);
  float* hbuf = (float*)(ws + h_off);
  unsigned* cnt = (unsigned*)(ws + bar_off);
  unsigned* gen = cnt + 1;

  const int smem = (64 * WROW + HH + 64 + 16) * 4;  // 267,584 B < 320 KB WGP LDS
  hipFuncSetAttribute((const void*)lstm_recur_kernel,
                      hipFuncAttributeMaxDynamicSharedMemorySize, smem);

  hipMemsetAsync(ws + bar_off, 0, 256, stream);  // reset barrier state each call (ws is poisoned)

  // 32x32 macro-tile per wave, 8 waves per block
  const int gemm_blocks = ((TT / 32) * (GG / 32)) / 8;  // 4096

  // ---- layer 0
  split_bf16_kernel<<<4096, 256, 0, stream>>>(x, xhi, xlo, TT * KK);
  split_bf16_kernel<<<4096, 256, 0, stream>>>(W_ih0, whi, wlo, GG * KK);
  wmma_gemm_bias_kernel<<<gemm_blocks, 256, 0, stream>>>(xhi, xlo, whi, wlo,
                                                         b_ih0, b_hh0, xg);
  lstm_recur_kernel<<<NWG, 256, smem, stream>>>(xg, W_hh0, y0, hbuf, nullptr, cnt, gen, TT);

  // ---- layer 1
  split_bf16_kernel<<<4096, 256, 0, stream>>>(y0, xhi, xlo, TT * KK);
  split_bf16_kernel<<<4096, 256, 0, stream>>>(W_ih1, whi, wlo, GG * KK);
  wmma_gemm_bias_kernel<<<gemm_blocks, 256, 0, stream>>>(xhi, xlo, whi, wlo,
                                                         b_ih1, b_hh1, xg);
  lstm_recur_kernel<<<NWG, 256, smem, stream>>>(xg, W_hh1, nullptr, hbuf, out, cnt, gen, TT);
}
